// M_11879879542696
// MI455X (gfx1250) — matrix operand order repaired and checked by reading, not compiled, not run on hardware
//
#include <hip/hip_runtime.h>
#include <hip/hip_bf16.h>

#define S   512
#define Dh  64
#define NH  12
#define HD  768
#define BHn 336

typedef __bf16 bf16;
typedef __attribute__((ext_vector_type(16))) __bf16 v16bf;
typedef __attribute__((ext_vector_type(8)))  __bf16 v8bf;
typedef __attribute__((ext_vector_type(8)))  float  v8f;

__constant__ const float kDROP  = 1.1111111111111112f; // 1/(1-0.1)
__constant__ const float kSCALE = 0.3535533905932738f; // 1/sqrt(8)

union V16u { v16bf v; v8bf h[2]; };

// WMMA A-operand (16x32 bf16): per-lane row, one contiguous 16-elem K run at [k0,k0+16)
__device__ __forceinline__ v16bf ldA(const bf16* row, int k0) {
  V16u u;
  u.h[0] = *(const v8bf*)(row + k0);
  u.h[1] = *(const v8bf*)(row + k0 + 8);
  return u.v;
}
// WMMA B-operand (32x16 bf16): per-lane column, one contiguous 16-elem K run
__device__ __forceinline__ v16bf ldB(const bf16* p) {
  V16u u;
  u.h[0] = *(const v8bf*)(p);
  u.h[1] = *(const v8bf*)(p + 8);
  return u.v;
}
__device__ __forceinline__ v8f wmma_bf16(v16bf a, v16bf b, v8f c) {
  return __builtin_amdgcn_wmma_f32_16x16x32_bf16(false, a, false, b, (short)0, c,
                                                 false, false);
}

// ---------------------------------------------------------------------------
// Kernel 1: dq = ((dOut@V^T)*mask*DROP*P  -  P * rowsum(...)) @ K  * SCALE
// Algebraic split: dq = dpm@K - diag(rowsum) * (P@K)   (single pass over P)
// grid = BH*4 blocks, 256 threads (8 waves); wave handles 16 rows.
// ---------------------------------------------------------------------------
__global__ __launch_bounds__(256, 1)
void k_dq(const bf16* __restrict__ dOut, const unsigned char* __restrict__ mask,
          const float* __restrict__ P, const bf16* __restrict__ VT,
          const bf16* __restrict__ Kmat, bf16* __restrict__ out_dq)
{
  // LDS: V^T transposed -> Vt[j][d] (pad 72), K transposed -> Kt[d][j] (pad 136),
  // per-wave bf16 A-staging scratch [16][40] x2 arrays.
  __shared__ __align__(16) bf16 Vt_lds[128 * 72];
  __shared__ __align__(16) bf16 Kt_lds[64 * 136];
  __shared__ __align__(16) bf16 scr[8 * 2 * 640];

  const int bh   = blockIdx.x >> 2;
  const int rb   = blockIdx.x & 3;
  const int tid  = threadIdx.x;
  const int lane = tid & 31, wv = tid >> 5;
  const int lo   = lane & 15, hi = lane >> 4;
  const int k0A  = hi ? 16 : 0;   // A-layout K base
  const int kbB  = hi ? 16 : 0;   // B-layout K base
  const int i0   = rb * 128 + wv * 16;

  const size_t pbase = (size_t)bh * S * S;
  const float* Prow = P + pbase;
  const unsigned char* Mrow = mask + pbase;

  // dOut A operands (k = d, 64 = 2 k-steps), kept in registers for all j
  const bf16* doutr = dOut + ((size_t)bh * S + i0 + lo) * Dh;
  v16bf a0 = ldA(doutr, k0A);
  v16bf a1 = ldA(doutr + 32, k0A);

  v8f dqA[4] = {};   // dpm @ K accumulators (64 d cols)
  v8f pkA[4] = {};   // P   @ K accumulators
  v8f rsum   = {};   // per-lane partial rowsums (fp32 dpm)

  bf16* my_dpm = scr + wv * 1280;        // [16][40] bf16
  bf16* my_p   = scr + wv * 1280 + 640;  // [16][40] bf16

  for (int ph = 0; ph < 4; ++ph) {
    const int J0 = ph * 128;
    // stage V^T (transposed) and K (transposed) for this 128-column band
    for (int idx = tid; idx < 128 * 64; idx += 256) {
      int d = idx >> 7, jr = idx & 127;
      Vt_lds[jr * 72 + d] = VT[((size_t)bh * Dh + d) * S + J0 + jr];
    }
    for (int idx = tid; idx < 128 * 64; idx += 256) {
      int jr = idx >> 6, d = idx & 63;
      Kt_lds[d * 136 + jr] = Kmat[((size_t)bh * S + J0 + jr) * Dh + d];
    }
    __syncthreads();

    for (int jj = 0; jj < 128; jj += 32) {
      // ---- dp tile [16 x 32] = dOut[16x64] @ V^T[64 x 32] ----
      const bf16* bv0 = Vt_lds + (jj + lo) * 72;
      const bf16* bv1 = Vt_lds + (jj + 16 + lo) * 72;
      v8f dp0 = {}, dp1 = {};
      dp0 = wmma_bf16(a0, ldB(bv0 + 0 + kbB), dp0);
      dp0 = wmma_bf16(a1, ldB(bv0 + 32 + kbB), dp0);
      dp1 = wmma_bf16(a0, ldB(bv1 + 0 + kbB), dp1);
      dp1 = wmma_bf16(a1, ldB(bv1 + 32 + kbB), dp1);

      // ---- elementwise: dpm = dp * mask * DROP * P ; rowsum partial ----
      const int jg0 = J0 + jj + lo;
#pragma unroll
      for (int r = 0; r < 8; ++r) {
        const int m = i0 + r + 8 * hi;
        const size_t off = (size_t)m * S + jg0;
        float p0 = Prow[off], p1 = Prow[off + 16];
        float s0 = Mrow[off] ? kDROP : 0.f;
        float s1 = Mrow[off + 16] ? kDROP : 0.f;
        float dm0 = dp0[r] * (p0 * s0);
        float dm1 = dp1[r] * (p1 * s1);
        rsum[r] += dm0 + dm1;
        const int mr = r + 8 * hi;
        my_dpm[mr * 40 + lo]      = (bf16)dm0;
        my_dpm[mr * 40 + 16 + lo] = (bf16)dm1;
        my_p[mr * 40 + lo]        = (bf16)p0;
        my_p[mr * 40 + 16 + lo]   = (bf16)p1;
      }
      __builtin_amdgcn_wave_barrier(); // same-wave LDS ops are in-order on CDNA5

      // re-materialize dpm / P tiles in WMMA A-layout from LDS staging
      v16bf aD = ldA(my_dpm + lo * 40, k0A);
      v16bf aP = ldA(my_p + lo * 40, k0A);

      // ---- accumulate dq-side GEMMs (k = j, 32 per iter) ----
#pragma unroll
      for (int dn = 0; dn < 4; ++dn) {
        const bf16* bk = Kt_lds + (dn * 16 + lo) * 136 + jj + kbB;
        v16bf b = ldB(bk);
        dqA[dn] = wmma_bf16(aD, b, dqA[dn]);
        pkA[dn] = wmma_bf16(aP, b, pkA[dn]);
      }
    }
    __syncthreads();
  }

  // reduce rowsums across the 16 lanes of each half-wave (wave32 shuffles)
#pragma unroll
  for (int r = 0; r < 8; ++r) {
    float v = rsum[r];
    v += __shfl_xor(v, 1, 32);
    v += __shfl_xor(v, 2, 32);
    v += __shfl_xor(v, 4, 32);
    v += __shfl_xor(v, 8, 32);
    rsum[r] = v;
  }

  const int bb = bh / NH, hh = bh % NH;
#pragma unroll
  for (int dn = 0; dn < 4; ++dn) {
#pragma unroll
    for (int r = 0; r < 8; ++r) {
      const int m = i0 + r + 8 * hi;
      float v = (dqA[dn][r] - rsum[r] * pkA[dn][r]) * kSCALE;
      out_dq[((size_t)(bb * S + m)) * HD + hh * Dh + dn * 16 + lo] = (bf16)v;
    }
  }
}

// ---------------------------------------------------------------------------
// Kernel 2: dv = (mask*P*DROP)^T @ dOut   (recompute p_drop, transpose via LDS)
// grid = BH*8 blocks (64-row dv chunks), 256 threads; 2 output tiles per wave.
// ---------------------------------------------------------------------------
__global__ __launch_bounds__(256, 1)
void k_dv(const bf16* __restrict__ dOut, const unsigned char* __restrict__ mask,
          const float* __restrict__ P, bf16* __restrict__ out_dv)
{
  __shared__ __align__(16) bf16 pdT[64 * 72];  // p_drop^T chunk [j][i]
  __shared__ __align__(16) bf16 dT[64 * 72];   // dOut^T chunk   [d][i]

  const int bh = blockIdx.x >> 3;
  const int ch = blockIdx.x & 7;
  const int j0 = ch * 64;
  const int tid = threadIdx.x;
  const int lane = tid & 31, wv = tid >> 5;
  const int lo = lane & 15, hi = lane >> 4;
  const int k0A = hi ? 16 : 0;
  const int kbB = hi ? 16 : 0;
  const int t0 = wv * 2;

  const size_t pbase = (size_t)bh * S * S;
  v8f acc[2] = {};

  for (int ic = 0; ic < S; ic += 64) {
    // coalesced global reads, transposed bf16 writes into LDS
    for (int idx = tid; idx < 64 * 64; idx += 256) {
      int ir = idx >> 6, jj = idx & 63;
      size_t off = pbase + (size_t)(ic + ir) * S + j0 + jj;
      float pd = mask[off] ? P[off] * kDROP : 0.f;
      pdT[jj * 72 + ir] = (bf16)pd;
    }
    for (int idx = tid; idx < 64 * 64; idx += 256) {
      int ir = idx >> 6, d = idx & 63;
      dT[d * 72 + ir] = dOut[((size_t)bh * S + ic + ir) * Dh + d];
    }
    __syncthreads();

#pragma unroll
    for (int t = 0; t < 2; ++t) {
      const int jm = (t0 + t) >> 2, dn = (t0 + t) & 3;
      const bf16* ar = pdT + (jm * 16 + lo) * 72;
      const bf16* br = dT + (dn * 16 + lo) * 72;
#pragma unroll
      for (int kk = 0; kk < 2; ++kk) {
        v16bf a = ldA(ar + kk * 32, k0A);
        v16bf b = ldB(br + kk * 32 + kbB);
        acc[t] = wmma_bf16(a, b, acc[t]);
      }
    }
    __syncthreads();
  }

  const int bb = bh / NH, hh = bh % NH;
#pragma unroll
  for (int t = 0; t < 2; ++t) {
    const int jm = (t0 + t) >> 2, dn = (t0 + t) & 3;
#pragma unroll
    for (int r = 0; r < 8; ++r) {
      const int j = j0 + jm * 16 + r + 8 * hi;
      out_dv[((size_t)(bb * S + j)) * HD + hh * Dh + dn * 16 + lo] =
          (bf16)acc[t][r];
    }
  }
}

// ---------------------------------------------------------------------------
extern "C" void kernel_launch(void* const* d_in, const int* in_sizes, int n_in,
                              void* d_out, int out_size, void* d_ws, size_t ws_size,
                              hipStream_t stream)
{
  (void)in_sizes; (void)n_in; (void)out_size; (void)d_ws; (void)ws_size;
  const bf16* dOut = (const bf16*)d_in[0];                 // clone_50  [B,H,S,D]
  const unsigned char* mask = (const unsigned char*)d_in[1]; // gt_scalar [B,H,S,S] bool
  const float* P = (const float*)d_in[2];                  // div_tensor [B,H,S,S]
  const bf16* VT = (const bf16*)d_in[3];                   // V^T [BH,D,S]
  // d_in[4] (Q^T) is dead code in the reference
  const bf16* Kmat = (const bf16*)d_in[5];                 // K [BH,S,D]

  bf16* out_dq = (bf16*)d_out;                              // [28,512,768]
  bf16* out_dv = out_dq + (size_t)28 * S * HD;              // [14336,768]

  k_dq<<<dim3(BHn * 4), dim3(256), 0, stream>>>(dOut, mask, P, VT, Kmat, out_dq);
  k_dv<<<dim3(BHn * 8), dim3(256), 0, stream>>>(dOut, mask, P, out_dv);
}
